// MultiHeadAttention_8100308321053
// MI455X (gfx1250) — matrix operand order repaired and checked by reading, not compiled, not run on hardware
//
#include <hip/hip_runtime.h>
#include <cstdint>

// ---------------------------------------------------------------------------
// Anchor attention for MI455X (gfx1250, wave32, WMMA + TDM).
// All 9 GEMM stages run through one templated bf16-WMMA tiled GEMM:
//   C(m,n) = alpha * sum_k op(A)(m,k)*op(B)(k,n) [+ bias(n)]
// op(A): !TA -> A[m*lda + k] ; TA -> A[k*lda + m]
// op(B): !TB -> B[k*ldb + n] ; TB -> B[n*ldb + k]
// Batched over (b,h) via blockIdx.z with independent base strides.
// bf16 K-contiguous operand tiles are staged by the Tensor Data Mover
// (tensor_load_to_lds, TENSORcnt); f32 / transposed tiles go through the
// VALU convert/transpose path.
// ---------------------------------------------------------------------------

typedef __attribute__((ext_vector_type(16))) __bf16    v16bf;
typedef __attribute__((ext_vector_type(8)))  __bf16    v8bf;
typedef __attribute__((ext_vector_type(8)))  float     v8f;
typedef __attribute__((ext_vector_type(8)))  uint16_t  u16x8;
typedef __attribute__((ext_vector_type(4)))  uint16_t  u16x4;
typedef __attribute__((ext_vector_type(4)))  float     f32x4;
typedef __attribute__((ext_vector_type(4)))  uint32_t  u32x4;
typedef __attribute__((ext_vector_type(8)))  int       i32x8;
typedef __attribute__((ext_vector_type(4)))  int       i32x4;

__device__ __forceinline__ uint16_t f32_to_bf16(float f) {
  uint32_t u = __builtin_bit_cast(uint32_t, f);
  uint32_t r = u + 0x7FFFu + ((u >> 16) & 1u);   // round-to-nearest-even
  return (uint16_t)(r >> 16);
}

constexpr int BM = 128;
constexpr int BN = 128;
constexpr int BK = 32;
constexpr int LDSROW = BK + 8;   // 40 bf16 = 80B row stride: 16B aligned, conflict-free

// ---------------------------------------------------------------------------
// TDM: DMA one 128-row x 32-col bf16 tile (row stride `ld` elements) from
// global memory into LDS at byte offset lds_off, inserting 16B of padding
// after every 64B row so the LDS image matches the [row][LDSROW] layout.
// Issued once per (designated) wave; completion via TENSORcnt.
// ---------------------------------------------------------------------------
__device__ __forceinline__ void tdm_load_tile_bf16(uint32_t lds_off,
                                                   const uint16_t* gptr,
                                                   int ld, int rows_valid)
{
  const uint64_t ga = (uint64_t)(uintptr_t)gptr;

  u32x4 g0;
  g0.x = 1u;                                   // count=1, user descriptor
  g0.y = lds_off;                              // lds_addr (bytes)
  g0.z = (uint32_t)ga;                         // global_addr[31:0]
  g0.w = (uint32_t)(ga >> 32) | (2u << 30);    // global_addr[56:32] | type=2

  const uint32_t td0 = 32u;                    // tensor_dim0 (elements)
  const uint32_t td1 = (uint32_t)rows_valid;   // tensor_dim1 (rows)
  const uint64_t st0 = (uint64_t)(uint32_t)ld; // tensor_dim0_stride (elements)

  i32x8 g1;
  g1[0] = (int)((1u << 16)    // data_size = 1 -> 2 bytes
              | (1u << 20)    // pad_enable
              | (3u << 22)    // pad_interval: 16 DWORDs (64B) between pads
              | (3u << 25));  // pad_amount:   4 DWORDs (16B)
  g1[1] = (int)((td0 & 0xFFFFu) << 16);                        // dim0[15:0]
  g1[2] = (int)(((td0 >> 16) & 0xFFFFu) | ((td1 & 0xFFFFu) << 16));
  g1[3] = (int)(((td1 >> 16) & 0xFFFFu) | (32u << 16));        // tile_dim0=32
  g1[4] = (int)(128u);                                         // tile_dim1=128
  g1[5] = (int)(uint32_t)st0;                                  // dim0_stride lo
  g1[6] = (int)(uint32_t)((st0 >> 32) & 0xFFFFu);              // dim0_stride hi
  g1[7] = 0;

  const i32x4 z4 = {0, 0, 0, 0};               // 2D tensor: groups 2/3 unused
  const i32x8 z8 = {0, 0, 0, 0, 0, 0, 0, 0};
  __builtin_amdgcn_tensor_load_to_lds(g0, g1, z4, z4, z8, 0);
}

// Stage one BM(or BN) x BK tile into LDS as [row][k] (k-contiguous), converting
// to bf16.  KCONTIG: memory is contiguous along k.  !KCONTIG: memory is
// contiguous along the row dim (transpose while staging).
template<typename T, bool KCONTIG>
__device__ __forceinline__ void stage_tile(uint16_t lds[][LDSROW],
                                           const T* __restrict__ g,
                                           long base, int ld,
                                           int r0, int k0, int Rmax, int tid)
{
  if constexpr (KCONTIG) {
#pragma unroll
    for (int i = 0; i < 4; ++i) {
      int c  = tid + i * 256;
      int r  = c >> 3;            // 8 chunks of 4 elems per 32-wide k row
      int kq = (c & 7) << 2;
      u16x4 v = {0, 0, 0, 0};
      if (r0 + r < Rmax) {
        const T* p = g + base + (long)(r0 + r) * ld + (k0 + kq);
        if constexpr (sizeof(T) == 4) {
          f32x4 f = *(const f32x4*)p;
          v.x = f32_to_bf16(f.x); v.y = f32_to_bf16(f.y);
          v.z = f32_to_bf16(f.z); v.w = f32_to_bf16(f.w);
        } else {
          v = *(const u16x4*)p;
        }
      }
      *(u16x4*)&lds[r][kq] = v;
    }
  } else {
#pragma unroll
    for (int i = 0; i < 4; ++i) {
      int c  = tid + i * 256;
      int kk = c >> 5;            // 32 chunks of 4 rows per k line
      int rq = (c & 31) << 2;
      uint16_t o[4];
      if (r0 + rq + 3 < Rmax) {
        const T* p = g + base + (long)(k0 + kk) * ld + (r0 + rq);
        if constexpr (sizeof(T) == 4) {
          f32x4 f = *(const f32x4*)p;
          o[0] = f32_to_bf16(f.x); o[1] = f32_to_bf16(f.y);
          o[2] = f32_to_bf16(f.z); o[3] = f32_to_bf16(f.w);
        } else {
          u16x4 u = *(const u16x4*)p;
          o[0] = u.x; o[1] = u.y; o[2] = u.z; o[3] = u.w;
        }
      } else {
#pragma unroll
        for (int j = 0; j < 4; ++j) {
          if (r0 + rq + j < Rmax) {
            const T* p = g + base + (long)(k0 + kk) * ld + (r0 + rq + j);
            if constexpr (sizeof(T) == 4) o[j] = f32_to_bf16(*p);
            else                          o[j] = (uint16_t)*p;
          } else o[j] = 0;
        }
      }
#pragma unroll
      for (int j = 0; j < 4; ++j) lds[rq + j][kk] = o[j];
    }
  }
}

// A fragment (16x32 MxK, bf16): lanes 0-15: M=lane, K = {0..7,16..23};
// lanes 16-31: M=lane-16, K = {8..15,24..31}.  Two aligned ds_load_b128.
__device__ __forceinline__ v16bf load_fragA(const uint16_t lds[][LDSROW],
                                            int mbase, int lane)
{
  int row = mbase + (lane & 15);
  int off = (lane < 16) ? 0 : 8;
  u16x8 lo = *(const u16x8*)&lds[row][off];
  u16x8 hi = *(const u16x8*)&lds[row][off + 16];
  v8bf l = __builtin_bit_cast(v8bf, lo);
  v8bf h = __builtin_bit_cast(v8bf, hi);
  return __builtin_shufflevector(l, h, 0,1,2,3,4,5,6,7,8,9,10,11,12,13,14,15);
}

// B fragment (32x16 KxN, bf16): lanes 0-15: N=lane, K = {0..7,16..23};
// lanes 16-31: N=lane-16, K = {8..15,24..31}.  LDS holds Bs[n][k].
__device__ __forceinline__ v16bf load_fragB(const uint16_t lds[][LDSROW],
                                            int nbase, int lane)
{
  int row = nbase + (lane & 15);
  int off = (lane < 16) ? 0 : 8;
  u16x8 lo = *(const u16x8*)&lds[row][off];
  u16x8 hi = *(const u16x8*)&lds[row][off + 16];
  v8bf l = __builtin_bit_cast(v8bf, lo);
  v8bf h = __builtin_bit_cast(v8bf, hi);
  return __builtin_shufflevector(l, h, 0,1,2,3,4,5,6,7,8,9,10,11,12,13,14,15);
}

template<typename AT, bool TA, typename BT, bool TB, bool OUT_BF16, bool BIAS>
__launch_bounds__(256)
__global__ void anchor_gemm(const AT* __restrict__ A, const BT* __restrict__ B,
                            const float* __restrict__ bias, void* __restrict__ Cv,
                            int Mdim, int Ndim, int Kdim,
                            long sAb, long sAh, int lda,
                            long sBb, long sBh, int ldb,
                            long sCb, long sCh, int ldc,
                            float alpha)
{
  __shared__ uint16_t As[2][BM][LDSROW];
  __shared__ uint16_t Bs[2][BN][LDSROW];

  // bf16 + K-contiguous tiles go through the Tensor Data Mover.
  constexpr bool TDMA = (sizeof(AT) == 2) && !TA;
  constexpr bool TDMB = (sizeof(BT) == 2) &&  TB;

  const int tid  = threadIdx.x;
  const int lane = tid & 31;
  const int wave = tid >> 5;
  const int wm   = wave >> 2;   // 0..1  -> 64 rows each
  const int wn   = wave & 3;    // 0..3  -> 32 cols each

  const int bh = blockIdx.z;
  const int b  = bh >> 3;
  const int h  = bh & 7;
  const long baseA = (long)b * sAb + (long)h * sAh;
  const long baseB = (long)b * sBb + (long)h * sBh;
  const long baseC = (long)b * sCb + (long)h * sCh;

  const int m0 = blockIdx.y * BM;
  const int n0 = blockIdx.x * BN;

  v8f acc[4][2] = {};

  const int nt = Kdim / BK;     // all K dims are multiples of 32

  auto stageA = [&](int buf, int kpos) {
    if constexpr (TDMA) {
      if (wave == 0)
        tdm_load_tile_bf16((uint32_t)(uintptr_t)&As[buf][0][0],
                           (const uint16_t*)A + baseA + (long)m0 * lda + kpos,
                           lda, Mdim - m0);
    } else {
      stage_tile<AT, !TA>(As[buf], A, baseA, lda, m0, kpos, Mdim, tid);
    }
  };
  auto stageB = [&](int buf, int kpos) {
    if constexpr (TDMB) {
      if (wave == 1)
        tdm_load_tile_bf16((uint32_t)(uintptr_t)&Bs[buf][0][0],
                           (const uint16_t*)B + baseB + (long)n0 * ldb + kpos,
                           ldb, Ndim - n0);
    } else {
      stage_tile<BT, TB>(Bs[buf], B, baseB, ldb, n0, kpos, Ndim, tid);
    }
  };
  auto tdm_fence = [&]() {
    if constexpr (TDMA || TDMB) {
      if ((TDMA && wave == 0) || (TDMB && wave == 1))
        __builtin_amdgcn_s_wait_tensorcnt(0);
    }
  };

  stageA(0, 0);
  stageB(0, 0);
  tdm_fence();
  __syncthreads();

  for (int t = 0; t < nt; ++t) {
    const int cur = t & 1;
    const int nxt = cur ^ 1;
    if (t + 1 < nt) {
      stageA(nxt, (t + 1) * BK);
      stageB(nxt, (t + 1) * BK);
    }

    v16bf af[4];
    v16bf bf[2];
#pragma unroll
    for (int r = 0; r < 4; ++r)
      af[r] = load_fragA(As[cur], wm * 64 + r * 16, lane);
#pragma unroll
    for (int c = 0; c < 2; ++c)
      bf[c] = load_fragB(Bs[cur], wn * 32 + c * 16, lane);

#pragma unroll
    for (int r = 0; r < 4; ++r)
#pragma unroll
      for (int c = 0; c < 2; ++c)
        acc[r][c] = __builtin_amdgcn_wmma_f32_16x16x32_bf16(
            false, af[r], false, bf[c], (short)0, acc[r][c], false, false);

    tdm_fence();
    __syncthreads();
  }

  // Epilogue: C/D layout: VGPR i: lanes 0-15 -> M=i, lanes 16-31 -> M=i+8;
  // N = lane % 16.
#pragma unroll
  for (int r = 0; r < 4; ++r) {
#pragma unroll
    for (int c = 0; c < 2; ++c) {
      const int col = n0 + wn * 32 + c * 16 + (lane & 15);
      if (col < Ndim) {
        float bv = 0.0f;
        if constexpr (BIAS) bv = bias[col];
#pragma unroll
        for (int i = 0; i < 8; ++i) {
          const int row = m0 + wm * 64 + r * 16 + ((lane < 16) ? i : i + 8);
          if (row < Mdim) {
            const float v = acc[r][c][i] * alpha + bv;
            const long idx = baseC + (long)row * ldc + col;
            if constexpr (OUT_BF16) ((uint16_t*)Cv)[idx] = f32_to_bf16(v);
            else                    ((float*)Cv)[idx]    = v;
          }
        }
      }
    }
  }
}

// ---------------------------------------------------------------------------

extern "C" void kernel_launch(void* const* d_in, const int* in_sizes, int n_in,
                              void* d_out, int out_size, void* d_ws, size_t ws_size,
                              hipStream_t stream)
{
  (void)in_sizes; (void)n_in; (void)out_size; (void)ws_size;

  const float* query = (const float*)d_in[0];
  const float* key   = (const float*)d_in[1];
  const float* value = (const float*)d_in[2];
  const float* Wq = (const float*)d_in[3];  const float* bq = (const float*)d_in[4];
  const float* Wk = (const float*)d_in[5];  const float* bk = (const float*)d_in[6];
  const float* Wv = (const float*)d_in[7];  const float* bv = (const float*)d_in[8];
  const float* Wa = (const float*)d_in[9];  const float* ba = (const float*)d_in[10];
  const float* Wo = (const float*)d_in[11]; const float* bo = (const float*)d_in[12];
  float* out = (float*)d_out;

  constexpr int B = 4, N = 2048, E = 512, H = 8, D = 64, Na = 512, Ea = 128;
  constexpr long NE = (long)N * E;                 // 1,048,576
  const float scale = 0.125f;                      // 1/sqrt(64)

  // bf16 workspace layout (elements)
  uint16_t* Qb  = (uint16_t*)d_ws;                 // [B*N, E]
  uint16_t* Kb  = Qb  + (long)B * N * E;
  uint16_t* Vb  = Kb  + (long)B * N * E;
  uint16_t* Ab  = Vb  + (long)B * N * E;           // [B*N, Ea] == [B, Na, H*D]
  uint16_t* KAb = Ab  + (long)B * N * Ea;          // [B,H,N,Na]
  uint16_t* QAb = KAb + (long)B * H * N * Na;      // [B,H,Na,N]
  uint16_t* ZTb = QAb + (long)B * H * Na * N;      // [B,H,Na,D]
  uint16_t* Ob  = ZTb + (long)B * H * Na * D;      // [B*N, E]

  dim3 blk(256);

  // ---- Stage 1: projections  (f32 in, bf16 out, bias) -----------------------
  {
    dim3 g(E / BN, (B * N) / BM, 1);
    anchor_gemm<float, false, float, false, true, true><<<g, blk, 0, stream>>>(
        query, Wq, bq, (void*)Qb, B * N, E, E,
        0L, 0L, E,  0L, 0L, E,  0L, 0L, E,  1.0f);
    anchor_gemm<float, false, float, false, true, true><<<g, blk, 0, stream>>>(
        key, Wk, bk, (void*)Kb, B * N, E, E,
        0L, 0L, E,  0L, 0L, E,  0L, 0L, E,  1.0f);
    anchor_gemm<float, false, float, false, true, true><<<g, blk, 0, stream>>>(
        value, Wv, bv, (void*)Vb, B * N, E, E,
        0L, 0L, E,  0L, 0L, E,  0L, 0L, E,  1.0f);
    dim3 ga(Ea / BN, (B * N) / BM, 1);
    anchor_gemm<float, false, float, false, true, true><<<ga, blk, 0, stream>>>(
        query, Wa, ba, (void*)Ab, B * N, Ea, E,
        0L, 0L, E,  0L, 0L, Ea,  0L, 0L, Ea,  1.0f);
  }

  // ---- Stage 2: k_a[b,h,n,m] = scale * K . A^T   (M=N, N=Na, K=D) ----------
  {
    dim3 g(Na / BN, N / BM, B * H);
    anchor_gemm<uint16_t, false, uint16_t, true, true, false><<<g, blk, 0, stream>>>(
        Kb, Ab, nullptr, (void*)KAb, N, Na, D,
        NE, (long)D, E,                       // K rows: b*NE + h*64 + n*E + k
        (long)N * Ea, (long)D, H * D,         // A rows: b*262144 + m*512 + h*64 + k
        (long)H * N * Na, (long)N * Na, Na,   // out [b,h,n,m]
        scale);
  }

  // ---- Stage 3: q_a[b,h,m,n] = scale * A . Q^T   (M=Na, N=N, K=D) ----------
  {
    dim3 g(N / BN, Na / BM, B * H);
    anchor_gemm<uint16_t, false, uint16_t, true, true, false><<<g, blk, 0, stream>>>(
        Ab, Qb, nullptr, (void*)QAb, Na, N, D,
        (long)N * Ea, (long)D, H * D,
        NE, (long)D, E,
        (long)H * Na * N, (long)Na * N, N,
        scale);
  }

  // ---- Stage 4: zT[b,h,m,dd] = scale * k_a^T . V  (M=Na, N=D, K=N) ---------
  {
    dim3 g(1, Na / BM, B * H);
    anchor_gemm<uint16_t, true, uint16_t, false, true, false><<<g, blk, 0, stream>>>(
        KAb, Vb, nullptr, (void*)ZTb, Na, D, N,
        (long)H * N * Na, (long)N * Na, Na,   // A(m,k)=KAb[... + k*Na + m]
        NE, (long)D, E,                       // B(k,n)=Vb[b*NE + h*64 + k*E + n]
        (long)H * Na * D, (long)Na * D, D,
        scale);
  }

  // ---- Stage 5: out[b,n,h*64+dd] = q_a^T . zT   (M=N, N=D, K=Na) -----------
  {
    dim3 g(1, N / BM, B * H);
    anchor_gemm<uint16_t, true, uint16_t, false, true, false><<<g, blk, 0, stream>>>(
        QAb, ZTb, nullptr, (void*)Ob, N, D, Na,
        (long)H * Na * N, (long)Na * N, N,    // A(m,k)=QAb[... + k*N + m]
        (long)H * Na * D, (long)Na * D, D,    // B(k,n)=ZTb[... + k*D + n]
        NE, (long)D, E,                       // C -> Ob[b*NE + m*E + h*64 + n]
        1.0f);
  }

  // ---- Stage 6: final = Ob . Wo + bo  (f32 out) -----------------------------
  {
    dim3 g(E / BN, (B * N) / BM, 1);
    anchor_gemm<uint16_t, false, float, false, false, true><<<g, blk, 0, stream>>>(
        Ob, Wo, bo, (void*)out, B * N, E, E,
        0L, 0L, E,  0L, 0L, E,  0L, 0L, E,  1.0f);
  }
}
